// Clip_OCR_Block_67499706024022
// MI455X (gfx1250) — matrix operand order repaired and checked by reading, not compiled, not run on hardware
//
#include <hip/hip_runtime.h>
#include <hip/hip_bf16.h>

// ---------------------------------------------------------------------------
// Problem constants (from reference setup_inputs)
// ---------------------------------------------------------------------------
#define N_BATCH 4
#define C_IN    512
#define KC_DIM  256
#define K_CLS   19
#define K_PAD   32      // classes padded to 32 for WMMA
#define S_HW    16384   // H*W = 128*128
#define OUT_CH  512

typedef __bf16 bf16_t;
typedef __attribute__((ext_vector_type(16))) __bf16 bf16x16;
typedef __attribute__((ext_vector_type(8)))  __bf16 bf16x8;
typedef __attribute__((ext_vector_type(4)))  __bf16 bf16x4;
typedef __attribute__((ext_vector_type(8)))  float  f32x8;
typedef __attribute__((ext_vector_type(4)))  float  f32x4;

// ---------------------------------------------------------------------------
// Helpers
// ---------------------------------------------------------------------------
__device__ __forceinline__ void store_y(float*  p, float v) { *p = v; }
__device__ __forceinline__ void store_y(bf16_t* p, float v) { *p = (bf16_t)v; }

// D = A(16x32 bf16) * B(32x16 bf16) + C(16x16 f32)
__device__ __forceinline__ f32x8 wmma_bf16(bf16x16 a, bf16x16 b, f32x8 c) {
  return __builtin_amdgcn_wmma_f32_16x16x32_bf16(
      /*neg_a=*/false, a, /*neg_b=*/false, b,
      /*c_mod=*/(short)0, c, /*reuse_a=*/false, /*reuse_b=*/false);
}

// A-fragment: lane holds row M=lane&15; lane-group g=lane>>4
// covers K = 16g..16g+15  -> two contiguous 8-half loads.
// B-fragment: lane holds col N=lane&15; group g covers K = 16g..16g+15 -> same
// helper with hi = lo + 8.
__device__ __forceinline__ bf16x16 ld_frag(const bf16_t* lo, const bf16_t* hi) {
  bf16x8 a = *(const bf16x8*)lo;
  bf16x8 b = *(const bf16x8*)hi;
  bf16x16 r;
#pragma unroll
  for (int i = 0; i < 8; ++i) { r[i] = a[i]; r[i + 8] = b[i]; }
  return r;
}

// ---------------------------------------------------------------------------
// Kernel 0: elementwise f32 -> bf16 (grid-stride, 4-wide). n must be %4 == 0.
// ---------------------------------------------------------------------------
__global__ __launch_bounds__(256)
void cvt_f32_bf16_kernel(const float* __restrict__ x, bf16_t* __restrict__ y,
                         long n4) {
  long i = (long)blockIdx.x * 256 + threadIdx.x;
  long stride = (long)gridDim.x * 256;
  for (; i < n4; i += stride) {
    f32x4 v = *(const f32x4*)(x + 4 * i);
    bf16x4 o;
#pragma unroll
    for (int j = 0; j < 4; ++j) o[j] = (bf16_t)v[j];
    *(bf16x4*)(y + 4 * i) = o;
  }
}

// ---------------------------------------------------------------------------
// Kernel 1: spatial softmax of probs -> bf16, padded to 32 class rows per n.
// grid.x = N_BATCH * 32; rows >= K_CLS are written as zeros so downstream WMMA
// can consume them unconditionally (branch-free proxy inner loop).
// ---------------------------------------------------------------------------
__global__ __launch_bounds__(256)
void softmax_spatial_kernel(const float* __restrict__ probs,
                            bf16_t* __restrict__ p) {
  const int S = S_HW;
  int b = blockIdx.x;
  int n = b >> 5, k = b & 31;
  int t = threadIdx.x;
  bf16_t* o = p + ((long)n * K_PAD + k) * S;
  if (k >= K_CLS) {                 // zero pad rows once
    for (int i = t; i < S; i += 256) o[i] = (bf16_t)0.f;
    return;
  }
  const float* x = probs + ((long)n * K_CLS + k) * S;
  __shared__ float red[256];
  float m = -3.4e38f;
  for (int i = t; i < S; i += 256) m = fmaxf(m, x[i]);
  red[t] = m; __syncthreads();
  for (int w = 128; w > 0; w >>= 1) { if (t < w) red[t] = fmaxf(red[t], red[t + w]); __syncthreads(); }
  m = red[0]; __syncthreads();
  float sum = 0.f;
  for (int i = t; i < S; i += 256) { float e = __expf(x[i] - m); sum += e; }
  red[t] = sum; __syncthreads();
  for (int w = 128; w > 0; w >>= 1) { if (t < w) red[t] += red[t + w]; __syncthreads(); }
  float inv = 1.f / red[0];
  for (int i = t; i < S; i += 256) o[i] = (bf16_t)(__expf(x[i] - m) * inv);
}

// ---------------------------------------------------------------------------
// Kernel 2: proxy[n][c][k] = sum_s featsb[n][c][s] * p[n][k][s]   (f32 out)
// WMMA GEMM, M=C tile 16, N=classes (32, zero-padded), Kdim = S (16384).
// grid: (C/128, N), block 256 = 8 waves, wave handles one 16-channel tile.
// p is zero-padded -> inner loop is branch-free b128 loads + 2 WMMAs.
// ---------------------------------------------------------------------------
__global__ __launch_bounds__(256)
void proxy_kernel(const bf16_t* __restrict__ feats, const bf16_t* __restrict__ p,
                  float* __restrict__ proxy) {
  const int S = S_HW;
  int t = threadIdx.x;
  int wave = t >> 5, lane = t & 31, g = lane >> 4, lr = lane & 15;
  int n = blockIdx.y;
  int c0 = blockIdx.x * 128 + wave * 16;
  const bf16_t* fn = feats + (long)n * C_IN * S;
  const bf16_t* pn = p + (long)n * K_PAD * S;
  f32x8 acc0 = {}, acc1 = {};
  const bf16_t* arow  = fn + (long)(c0 + lr) * S;
  const bf16_t* b0row = pn + (long)lr * S;
  const bf16_t* b1row = pn + (long)(16 + lr) * S;   // zero rows for cls >= 19
  for (int sc = 0; sc < S; sc += 32) {
    bf16x16 a  = ld_frag(arow + sc + 16 * g, arow + sc + 16 * g + 8);
    bf16x16 b0 = ld_frag(b0row + sc + 16 * g, b0row + sc + 16 * g + 8);
    bf16x16 b1 = ld_frag(b1row + sc + 16 * g, b1row + sc + 16 * g + 8);
    acc0 = wmma_bf16(a, b0, acc0);
    acc1 = wmma_bf16(a, b1, acc1);
  }
  float* pr = proxy + (long)n * C_IN * K_CLS;
  bool hasB1 = (lr < (K_CLS - 16));
#pragma unroll
  for (int r = 0; r < 8; ++r) {
    int c = c0 + r + 8 * g;
    pr[(long)c * K_CLS + lr] = acc0[r];
    if (hasB1) pr[(long)c * K_CLS + 16 + lr] = acc1[r];
  }
}

// ---------------------------------------------------------------------------
// Kernel 3: tiny conv1x1+BN+ReLU on proxies (19 columns). Plain VALU — this is
// bandwidth-trivial. Optionally writes class-major (transpose) and zero-pads
// classes [19,Kpad) so the attention kernel can consume padded 32-class tiles.
// ---------------------------------------------------------------------------
template <typename TO>
__global__ __launch_bounds__(256)
void small_cbr_kernel(const float* __restrict__ W, const float* __restrict__ bias,
                      const float* __restrict__ scale, const float* __restrict__ shift,
                      const float* __restrict__ X, TO* __restrict__ out,
                      int O, int Cin, int Kpad, int ldx, int transpose, int ldout,
                      long xStrideN, long oStrideN) {
  int n = blockIdx.y;
  int tid = blockIdx.x * 256 + threadIdx.x;
  if (tid >= O * Kpad) return;
  int k = tid % Kpad, o = tid / Kpad;
  const float* x = X + (long)n * xStrideN;
  TO* y = out + (long)n * oStrideN;
  float v = 0.f;
  if (k < K_CLS) {
    float acc = 0.f;
    const float* wr = W + (long)o * Cin;
#pragma unroll 4
    for (int c = 0; c < Cin; ++c) acc += wr[c] * x[(long)c * ldx + k];
    v = (acc + bias[o]) * scale[o] + shift[o];
    v = v > 0.f ? v : 0.f;
  }
  long idx = transpose ? (long)k * ldout + o : (long)o * ldout + k;
  store_y(&y[idx], v);
}

// ---------------------------------------------------------------------------
// Kernel 4: generic fused per-pixel GEMM: Y[o][s] = relu((W[o,:]·X[:,s]+b)*sc+sh)
// 64x64 block tile, double-buffered LDS (single barrier per k-step): tile i+1
// is loaded global->regs before the barrier and stored to the alternate LDS
// buffer after it, overlapping HBM latency with WMMA on tile i.
// Ksplit implements concat([X1;X2]) along channels without materializing it.
// ---------------------------------------------------------------------------
template <typename TY>
__global__ __launch_bounds__(256)
void gemm_cbr_kernel(const bf16_t* __restrict__ W, const float* __restrict__ bias,
                     const float* __restrict__ scale, const float* __restrict__ shift,
                     const bf16_t* __restrict__ X1, const bf16_t* __restrict__ X2,
                     TY* __restrict__ Y,
                     int K, int Ksplit,
                     long xStride1, long xStride2, long yStride) {
  const int S = S_HW;
  __shared__ __align__(16) bf16_t ldsA[2][64 * 40];  // [o_local][k], stride 40
  __shared__ __align__(16) bf16_t ldsB[2][64 * 40];  // [s_local][k] (transposed)

  const int t = threadIdx.x;
  const int wave = t >> 5, lane = t & 31;
  const int g = lane >> 4, lr = lane & 15;
  const int mw = (wave >> 1) * 16;   // M sub-tile: 0/16/32/48
  const int nw = (wave & 1) * 32;    // N sub-tile: 0/32
  const int s0 = blockIdx.x * 64;
  const int o0 = blockIdx.y * 64;
  const int n  = blockIdx.z;

  const bf16_t* x1 = X1 + (long)n * xStride1;
  const bf16_t* x2 = X2 + (long)n * xStride2;
  TY* y = Y + (long)n * yStride;

  f32x8 acc0 = {}, acc1 = {};
  const int arow = t >> 2, akc = (t & 3) * 8;   // A stage: 64 rows x 32 k
  const int brow = t >> 3, bsc = (t & 7) * 8;   // B stage: 32 k x 64 s
  const bf16_t* aSrc = W + (long)(o0 + arow) * K + akc;
  auto bSrc = [&](int i) -> const bf16_t* {
    int kg = i * 32 + brow;
    return (kg < Ksplit) ? x1 + (long)kg * S + s0 + bsc
                         : x2 + (long)(kg - Ksplit) * S + s0 + bsc;
  };

  // prologue: stage tile 0
  bf16x8 aReg = *(const bf16x8*)aSrc;
  bf16x8 bReg = *(const bf16x8*)bSrc(0);
  *(bf16x8*)&ldsA[0][arow * 40 + akc] = aReg;
#pragma unroll
  for (int j = 0; j < 8; ++j) ldsB[0][(bsc + j) * 40 + brow] = bReg[j];

  const int nk = K >> 5;
  for (int i = 0; i < nk; ++i) {
    const int cur = i & 1;
    if (i + 1 < nk) {                       // global -> regs for tile i+1
      aReg = *(const bf16x8*)(aSrc + (long)(i + 1) * 32);
      bReg = *(const bf16x8*)bSrc(i + 1);
      if (i + 2 < nk) __builtin_prefetch(aSrc + (long)(i + 2) * 32, 0, 0);
    }
    __syncthreads();                        // tile i fully in LDS
    bf16x16 a  = ld_frag(&ldsA[cur][(mw + lr) * 40 + 16 * g],
                         &ldsA[cur][(mw + lr) * 40 + 16 * g + 8]);
    bf16x16 b0 = ld_frag(&ldsB[cur][(nw + lr) * 40 + 16 * g],
                         &ldsB[cur][(nw + lr) * 40 + 16 * g + 8]);
    bf16x16 b1 = ld_frag(&ldsB[cur][(nw + 16 + lr) * 40 + 16 * g],
                         &ldsB[cur][(nw + 16 + lr) * 40 + 16 * g + 8]);
    acc0 = wmma_bf16(a, b0, acc0);
    acc1 = wmma_bf16(a, b1, acc1);
    if (i + 1 < nk) {                       // regs -> alternate LDS buffer
      const int nxt = cur ^ 1;
      *(bf16x8*)&ldsA[nxt][arow * 40 + akc] = aReg;
#pragma unroll
      for (int j = 0; j < 8; ++j) ldsB[nxt][(bsc + j) * 40 + brow] = bReg[j];
    }
  }

#pragma unroll
  for (int r = 0; r < 8; ++r) {
    int o = o0 + mw + r + 8 * g;
    float bb = bias[o], ss = scale[o], tt = shift[o];
    float v0 = (acc0[r] + bb) * ss + tt; v0 = v0 > 0.f ? v0 : 0.f;
    float v1 = (acc1[r] + bb) * ss + tt; v1 = v1 > 0.f ? v1 : 0.f;
    store_y(&y[(long)o * S + s0 + nw + lr], v0);
    store_y(&y[(long)o * S + s0 + nw + 16 + lr], v1);
  }
}

// ---------------------------------------------------------------------------
// Kernel 5: attention. Per 64-pixel block: logits = kk^T·q (WMMA over KC=256),
// softmax over 19 classes, ctx = val^T·sim (one WMMA, Kdim padded to 32).
// kkT: (32cls x 256) class-major bf16, zero-padded.
// valC: (256 x 32cls) chan-major bf16, zero-padded.
// ---------------------------------------------------------------------------
__global__ __launch_bounds__(256)
void attention_kernel(const bf16_t* __restrict__ q, const bf16_t* __restrict__ kkT,
                      const bf16_t* __restrict__ valC, bf16_t* __restrict__ ctx) {
  const int S = S_HW, KC = KC_DIM;
  __shared__ __align__(16) bf16_t ldsQ[64 * 264];   // [pixel][c]
  __shared__ __align__(16) bf16_t ldsK[32 * 264];   // [class][c]
  __shared__ __align__(16) bf16_t ldsV[256 * 40];   // [c][class]
  __shared__ __align__(16) float  ldsL[64 * 32];    // [pixel][class] logits
  __shared__ __align__(16) bf16_t ldsSim[64 * 32];  // [pixel][class] softmax

  int t = threadIdx.x;
  int wave = t >> 5, lane = t & 31, g = lane >> 4, lr = lane & 15;
  int n = blockIdx.y, s0 = blockIdx.x * 64;

  const bf16_t* qn = q + (long)n * KC * S;
  for (int i = t; i < KC * 64; i += 256) {          // transpose q tile into LDS
    int c = i >> 6, s = i & 63;
    ldsQ[s * 264 + c] = qn[(long)c * S + s0 + s];
  }
  const bf16_t* kn = kkT + (long)n * 32 * KC;
  for (int i = t; i < 32 * KC; i += 256) {
    int cls = i >> 8, c = i & 255;
    ldsK[cls * 264 + c] = kn[i];
  }
  const bf16_t* vn = valC + (long)n * KC * 32;
  for (int i = t; i < KC * 32; i += 256) {
    int c = i >> 5, cls = i & 31;
    ldsV[c * 40 + cls] = vn[i];
  }
  __syncthreads();

  {   // logits: (32 class) x (64 pixel), Kdim = 256
    int mt = wave >> 2, pt = wave & 3;
    f32x8 acc = {};
    for (int kc = 0; kc < KC; kc += 32) {
      bf16x16 a = ld_frag(&ldsK[(mt * 16 + lr) * 264 + kc + 16 * g],
                          &ldsK[(mt * 16 + lr) * 264 + kc + 16 * g + 8]);
      bf16x16 b = ld_frag(&ldsQ[(pt * 16 + lr) * 264 + kc + 16 * g],
                          &ldsQ[(pt * 16 + lr) * 264 + kc + 16 * g + 8]);
      acc = wmma_bf16(a, b, acc);
    }
#pragma unroll
    for (int r = 0; r < 8; ++r)
      ldsL[(pt * 16 + lr) * 32 + mt * 16 + r + 8 * g] = acc[r];
  }
  __syncthreads();

  if (t < 64) {   // per-pixel softmax over 19 classes (scaled by KC^-0.5)
    float sc = rsqrtf((float)KC);
    float l[K_CLS];
    float m = -3.4e38f;
#pragma unroll
    for (int k = 0; k < K_CLS; ++k) { l[k] = ldsL[t * 32 + k] * sc; m = fmaxf(m, l[k]); }
    float sum = 0.f;
#pragma unroll
    for (int k = 0; k < K_CLS; ++k) { l[k] = __expf(l[k] - m); sum += l[k]; }
    float inv = 1.f / sum;
#pragma unroll
    for (int k = 0; k < K_CLS; ++k) ldsSim[t * 32 + k] = (bf16_t)(l[k] * inv);
#pragma unroll
    for (int k = K_CLS; k < 32; ++k) ldsSim[t * 32 + k] = (bf16_t)0.f;
  }
  __syncthreads();

  // ctx: (256 c) x (64 pixel), Kdim = 32 -> one WMMA per 16x16 tile
  bf16_t* cn = ctx + (long)n * KC * S;
#pragma unroll
  for (int i = 0; i < 8; ++i) {
    int tile = wave + i * 8;
    int mt = tile >> 2, pt = tile & 3;
    bf16x16 a = ld_frag(&ldsV[(mt * 16 + lr) * 40 + 16 * g],
                        &ldsV[(mt * 16 + lr) * 40 + 16 * g + 8]);
    bf16x16 b = ld_frag(&ldsSim[(pt * 16 + lr) * 32 + 16 * g],
                        &ldsSim[(pt * 16 + lr) * 32 + 16 * g + 8]);
    f32x8 z = {};
    f32x8 d = wmma_bf16(a, b, z);
#pragma unroll
    for (int r = 0; r < 8; ++r)
      cn[(long)(mt * 16 + r + 8 * g) * S + s0 + pt * 16 + lr] = (bf16_t)d[r];
  }
}

// ---------------------------------------------------------------------------
// Host launcher
// ---------------------------------------------------------------------------
extern "C" void kernel_launch(void* const* d_in, const int* in_sizes, int n_in,
                              void* d_out, int out_size, void* d_ws, size_t ws_size,
                              hipStream_t stream) {
  (void)in_sizes; (void)n_in; (void)out_size; (void)ws_size;
  const float* feats = (const float*)d_in[0];
  const float* probs = (const float*)d_in[1];
  const float* wp1 = (const float*)d_in[2];  const float* bp1 = (const float*)d_in[3];
  const float* sp1 = (const float*)d_in[4];  const float* tp1 = (const float*)d_in[5];
  const float* wp2 = (const float*)d_in[6];  const float* bp2 = (const float*)d_in[7];
  const float* sp2 = (const float*)d_in[8];  const float* tp2 = (const float*)d_in[9];
  const float* wo1 = (const float*)d_in[10]; const float* bo1 = (const float*)d_in[11];
  const float* so1 = (const float*)d_in[12]; const float* to1 = (const float*)d_in[13];
  const float* wo2 = (const float*)d_in[14]; const float* bo2 = (const float*)d_in[15];
  const float* so2 = (const float*)d_in[16]; const float* to2 = (const float*)d_in[17];
  const float* wdw = (const float*)d_in[18]; const float* bdw = (const float*)d_in[19];
  const float* sdw = (const float*)d_in[20]; const float* tdw = (const float*)d_in[21];
  const float* wu  = (const float*)d_in[22]; const float* bu  = (const float*)d_in[23];
  const float* su  = (const float*)d_in[24]; const float* tu  = (const float*)d_in[25];
  const float* wf  = (const float*)d_in[26]; const float* bff = (const float*)d_in[27];
  const float* sfc = (const float*)d_in[28]; const float* tfc = (const float*)d_in[29];
  // d_in[30] = clip_num (0, identity) — unused.

  float* out = (float*)d_out;
  char* ws = (char*)d_ws;
  size_t off = 0;
  auto take = [&](size_t bytes) -> void* {
    size_t cur = (off + 255) & ~(size_t)255;
    off = cur + bytes;
    return (void*)(ws + cur);
  };

  const long S = S_HW;
  bf16_t* featsb = (bf16_t*)take((size_t)N_BATCH * C_IN * S * 2);
  bf16_t* wp1b   = (bf16_t*)take((size_t)KC_DIM * C_IN * 2);
  bf16_t* wp2b   = (bf16_t*)take((size_t)KC_DIM * KC_DIM * 2);
  bf16_t* wub    = (bf16_t*)take((size_t)C_IN * KC_DIM * 2);
  bf16_t* wfb    = (bf16_t*)take((size_t)OUT_CH * 2 * C_IN * 2);
  bf16_t* p_sm   = (bf16_t*)take((size_t)N_BATCH * K_PAD * S * 2);   // padded
  float*  proxy  = (float*) take((size_t)N_BATCH * C_IN * K_CLS * 4);
  float*  t1     = (float*) take((size_t)N_BATCH * KC_DIM * K_CLS * 4);
  bf16_t* kkT    = (bf16_t*)take((size_t)N_BATCH * K_PAD * KC_DIM * 2);
  bf16_t* valC   = (bf16_t*)take((size_t)N_BATCH * KC_DIM * K_PAD * 2);
  bf16_t* tmp1   = (bf16_t*)take((size_t)N_BATCH * KC_DIM * S * 2);
  bf16_t* qb     = (bf16_t*)take((size_t)N_BATCH * KC_DIM * S * 2);
  bf16_t* upb    = (bf16_t*)take((size_t)N_BATCH * C_IN * S * 2);
  bf16_t* ctxb   = tmp1;   // tmp1 dead after conv2 -> reuse for ctx

  dim3 blk(256);

  // ---- precision pre-pass: f32 -> bf16 for feats and big weights ----------
  cvt_f32_bf16_kernel<<<dim3(4096), blk, 0, stream>>>(
      feats, featsb, (long)N_BATCH * C_IN * S / 4);
  cvt_f32_bf16_kernel<<<dim3(128), blk, 0, stream>>>(
      wp1, wp1b, (long)KC_DIM * C_IN / 4);
  cvt_f32_bf16_kernel<<<dim3(64), blk, 0, stream>>>(
      wp2, wp2b, (long)KC_DIM * KC_DIM / 4);
  cvt_f32_bf16_kernel<<<dim3(128), blk, 0, stream>>>(
      wu, wub, (long)C_IN * KC_DIM / 4);
  cvt_f32_bf16_kernel<<<dim3(512), blk, 0, stream>>>(
      wf, wfb, (long)OUT_CH * 2 * C_IN / 4);

  softmax_spatial_kernel<<<dim3(N_BATCH * K_PAD), blk, 0, stream>>>(probs, p_sm);

  proxy_kernel<<<dim3(C_IN / 128, N_BATCH), blk, 0, stream>>>(featsb, p_sm, proxy);

  // f_object stage 1: t1 = cbr(wo1 . proxy)           (256 x 19, f32)
  small_cbr_kernel<float><<<dim3((KC_DIM * K_CLS + 255) / 256, N_BATCH), blk, 0, stream>>>(
      wo1, bo1, so1, to1, proxy, t1, KC_DIM, C_IN, K_CLS, K_CLS, 0, K_CLS,
      (long)C_IN * K_CLS, (long)KC_DIM * K_CLS);
  // f_object stage 2: kkT = cbr(wo2 . t1), class-major bf16, zero pad to 32 rows
  small_cbr_kernel<bf16_t><<<dim3((KC_DIM * K_PAD + 255) / 256, N_BATCH), blk, 0, stream>>>(
      wo2, bo2, so2, to2, t1, kkT, KC_DIM, KC_DIM, K_PAD, K_CLS, 1, KC_DIM,
      (long)KC_DIM * K_CLS, (long)K_PAD * KC_DIM);
  // f_down: valC = cbr(wd . proxy), chan-major bf16, zero pad to 32 class cols
  small_cbr_kernel<bf16_t><<<dim3((KC_DIM * K_PAD + 255) / 256, N_BATCH), blk, 0, stream>>>(
      wdw, bdw, sdw, tdw, proxy, valC, KC_DIM, C_IN, K_PAD, K_CLS, 0, K_PAD,
      (long)C_IN * K_CLS, (long)KC_DIM * K_PAD);

  // f_pixel conv1: tmp1 = cbr(wp1 . feats)
  gemm_cbr_kernel<bf16_t>
      <<<dim3(S_HW / 64, KC_DIM / 64, N_BATCH), blk, 0, stream>>>(
      wp1b, bp1, sp1, tp1, featsb, featsb, tmp1, C_IN, C_IN,
      (long)C_IN * S, (long)C_IN * S, (long)KC_DIM * S);
  // f_pixel conv2: qb = cbr(wp2 . tmp1)
  gemm_cbr_kernel<bf16_t>
      <<<dim3(S_HW / 64, KC_DIM / 64, N_BATCH), blk, 0, stream>>>(
      wp2b, bp2, sp2, tp2, tmp1, featsb, qb, KC_DIM, KC_DIM,
      (long)KC_DIM * S, (long)C_IN * S, (long)KC_DIM * S);

  attention_kernel<<<dim3(S_HW / 64, N_BATCH), blk, 0, stream>>>(qb, kkT, valC, ctxb);

  // f_up: upb = cbr(wu . ctx)
  gemm_cbr_kernel<bf16_t>
      <<<dim3(S_HW / 64, C_IN / 64, N_BATCH), blk, 0, stream>>>(
      wub, bu, su, tu, ctxb, featsb, upb, KC_DIM, KC_DIM,
      (long)KC_DIM * S, (long)C_IN * S, (long)C_IN * S);

  // final fusion: out = cbr(wf . concat([up, feats]))  (Ksplit = 512)
  gemm_cbr_kernel<float>
      <<<dim3(S_HW / 64, OUT_CH / 64, N_BATCH), blk, 0, stream>>>(
      wfb, bff, sfc, tfc, upb, featsb, out, 2 * C_IN, C_IN,
      (long)C_IN * S, (long)C_IN * S, (long)OUT_CH * S);
}
